// BertClassifier_37907381354988
// MI455X (gfx1250) — matrix operand order repaired and checked by reading, not compiled, not run on hardware
//
#include <hip/hip_runtime.h>
#include <hip/hip_bf16.h>

typedef __attribute__((ext_vector_type(16))) _Float16 v16h;
typedef __attribute__((ext_vector_type(8)))  float    v8f;

#define Mx   253
#define Bx   2
#define Lx   256
#define Hx   768
#define D1x  770
#define OUTx 40
#define FIDx 700
#define D1P  800            // K padded to 25*32
#define KC2  25             // K chunks (32 each) in span GEMM
#define RPAD 512            // B*M = 506 rows padded to 512 (unconditional tile stores)
// ws layout in floats
#define AI_OFF  0                            // 512*800 = 409600
#define AJ_OFF  (RPAD*D1P)                   // 409600
#define B1_OFF  (2*RPAD*D1P)                 // 819200
#define WI_OFF  (B1_OFF + D1P)               // 820000
#define W2F_OFF (WI_OFF + D1P)               // 820800 (halves region)
#define W2F_HALVES (KC2*3*32*16)             // 38400 halves = 19200 floats
#define W1F_OFF (W2F_OFF + W2F_HALVES/2)     // 840000 (halves region)
#define W1F_TILES (100*24)                   // 100 col tiles x 24 K chunks
#define W1F_HALVES (W1F_TILES*32*16)         // 1,228,800 halves
// total ws: (840000 + 614400) floats = ~5.82 MB

// ---- prep: pad b1 / w_ind; pre-swizzle W2 and W1 into per-lane WMMA B-fragments ----
__global__ void prep_kernel(const float* __restrict__ W1, const float* __restrict__ b1,
                            const float* __restrict__ W2, float* __restrict__ ws) {
    int t = blockIdx.x * blockDim.x + threadIdx.x;
    float* b1p = ws + B1_OFF;
    float* wip = ws + WI_OFF;
    _Float16* w2f = (_Float16*)(ws + W2F_OFF);
    _Float16* w1f = (_Float16*)(ws + W1F_OFF);
    if (t < D1P) {
        b1p[t] = (t < D1x) ? b1[t] : 0.0f;
        wip[t] = (t < D1x) ? W1[(size_t)(2 * Hx) * D1x + t] : 0.0f;   // W1[2H] row
    }
    if (t < W2F_HALVES) {
        int i    = t & 15;
        int lane = (t >> 4) & 31;
        int tile = t >> 9;            // kc*3 + nt
        int nt = tile % 3, kc = tile / 3;
        int K = kc * 32 + ((lane < 16) ? 0 : 8) + i + ((i < 8) ? 0 : 8);    // B layout: lanes 0-15 K=0-7,16-23; 16-31 K=8-15,24-31
        int N = nt * 16 + (lane & 15);
        float v = (K < D1x && N < OUTx) ? W2[(size_t)K * OUTx + N] : 0.0f;
        w2f[t] = (_Float16)v;
    }
    if (t < W1F_HALVES) {             // layout: [ct(100)][kc(24)][lane(32)][i(16)]
        int i    = t & 15;
        int lane = (t >> 4) & 31;
        int tile = t >> 9;            // ct*24 + kc
        int kc = tile % 24, ct = tile / 24;
        int half = ct / 50;           // 0 = Ai weights, 1 = Aj weights
        int K = kc * 32 + ((lane < 16) ? 0 : 8) + i + ((i < 8) ? 0 : 8);    // K < 768 always
        int d = (ct % 50) * 16 + (lane & 15);            // output column (0..799)
        float v = (d < D1x) ? W1[(size_t)(half * Hx + K) * D1x + d] : 0.0f;
        w1f[t] = (_Float16)v;
    }
}

// ---- Ai/Aj = tok @ W1[:H] / W1[H:2H]; f32 out stride D1P; rows 506..511 are scratch ----
__global__ void aiaj_kernel(const float* __restrict__ hidden, const float* __restrict__ ws_c,
                            float* __restrict__ ws) {
    int lane = threadIdx.x;
    int bt = blockIdx.x;            // row tile 0..31 (rows 0..511)
    int ct = blockIdx.y;            // col tile 0..99 (50 for Ai, 50 for Aj; cols to 800)
    int half = ct / 50;
    int d0 = (ct % 50) * 16;
    float* dst = ws + (half ? AJ_OFF : AI_OFF);
    const v16h* w1f = (const v16h*)(ws_c + W1F_OFF);

    int mloc = lane & 15;
    int kbA = (lane < 16) ? 0 : 8;      // 16-bit A-matrix layout interleave
    int rowIdx = bt * 16 + mloc;
    int rc = rowIdx > (Bx * Mx - 1) ? (Bx * Mx - 1) : rowIdx;   // clamp reads for pad rows
    const float* arow = hidden + ((size_t)((rc / Mx) * Lx + 1 + (rc % Mx))) * Hx;

    v8f c = {};
    for (int kc = 0; kc < 24; ++kc) {           // K = 768 exactly
        int k0 = kc * 32;
        v16h a;
#pragma unroll
        for (int i = 0; i < 8; ++i) {
            a[i]     = (_Float16)arow[k0 + kbA + i];
            a[8 + i] = (_Float16)arow[k0 + 16 + kbA + i];
        }
        v16h bf = w1f[(ct * 24 + kc) * 32 + lane];
        c = __builtin_amdgcn_wmma_f32_16x16x32_f16(false, a, false, bf, (short)0, c, false, false);
    }
    // D layout: VGPR r -> row (bt*16 + r + 8*(lane>=16)); ob*Mx+om == row, so index directly.
    int rofs = (lane < 16) ? 0 : 8;
    int ncol = lane & 15;
    float* dp = dst + (size_t)(bt * 16 + rofs) * D1P + d0 + ncol;
#pragma unroll
    for (int r = 0; r < 8; ++r)
        dp[(size_t)r * D1P] = c[r];
}

// ---- fused span head: h=relu(Ai[ii]+Aj[jj]+ind*w_ind+b1); logits=h@W2; log_softmax; mask ----
__global__ void span_kernel(const float* __restrict__ ws, const float* __restrict__ b2g,
                            const int* __restrict__ pred_spans, const int* __restrict__ span_avail,
                            float* __restrict__ out) {
    __shared__ float tile[16 * 48];
    int lane = threadIdx.x;
    int jj0 = blockIdx.x * 16;      // 16 tiles cover jj 0..255 (253 valid)
    int ii  = blockIdx.y;           // 0..252
    int b   = blockIdx.z;           // 0..1

    const float* Ai  = ws + AI_OFF;
    const float* Aj  = ws + AJ_OFF;
    const float* b1p = ws + B1_OFF;
    const float* wip = ws + WI_OFF;
    const v16h*  w2f = (const v16h*)(ws + W2F_OFF);

    int s = pred_spans[b * 2], e = pred_spans[b * 2 + 1];
    int mloc = lane & 15;
    int jj = jj0 + mloc;
    int jjc = jj > (Mx - 1) ? (Mx - 1) : jj;
    float ind = 0.0f;
    if (ii == s && jj == e) ind = 2.0f;
    else if (s <= ii && jj <= e && ii <= jj) ind = 1.0f;

    const float* ai = Ai + (size_t)(b * Mx + ii) * D1P;
    const float* aj = Aj + (size_t)(b * Mx + jjc) * D1P;
    int kb = (lane < 16) ? 0 : 8;

    v8f c0 = {}, c1 = {}, c2 = {};
    for (int kc = 0; kc < KC2; ++kc) {
        int k0 = kc * 32;
        v16h a;
#pragma unroll
        for (int i = 0; i < 8; ++i) {
            int k = k0 + kb + i;
            float h = ai[k] + aj[k] + fmaf(ind, wip[k], b1p[k]);
            a[i] = (_Float16)fmaxf(h, 0.0f);
            k = k0 + 16 + kb + i;
            h = ai[k] + aj[k] + fmaf(ind, wip[k], b1p[k]);
            a[8 + i] = (_Float16)fmaxf(h, 0.0f);
        }
        v16h bf0 = w2f[(kc * 3 + 0) * 32 + lane];
        v16h bf1 = w2f[(kc * 3 + 1) * 32 + lane];
        v16h bf2 = w2f[(kc * 3 + 2) * 32 + lane];
        c0 = __builtin_amdgcn_wmma_f32_16x16x32_f16(false, a, false, bf0, (short)0, c0, false, false);
        c1 = __builtin_amdgcn_wmma_f32_16x16x32_f16(false, a, false, bf1, (short)0, c1, false, false);
        c2 = __builtin_amdgcn_wmma_f32_16x16x32_f16(false, a, false, bf2, (short)0, c2, false, false);
    }
    // D layout: VGPR r, lanes 0-15 -> (M=r, N=lane), lanes 16-31 -> (M=r+8, N=lane-16)
    int rofs = (lane < 16) ? 0 : 8;
    int ncol = lane & 15;
#pragma unroll
    for (int r = 0; r < 8; ++r) {
        tile[(r + rofs) * 48 + ncol]      = c0[r];
        tile[(r + rofs) * 48 + 16 + ncol] = c1[r];
        tile[(r + rofs) * 48 + 32 + ncol] = c2[r];
    }
    __syncthreads();
    if (lane < 16) {
        int jr = jj0 + lane;
        if (jr < Mx) {
            const float4* row4 = (const float4*)(tile + lane * 48);
            const float4* b24  = (const float4*)b2g;
            float mx = -3.4e38f;
#pragma unroll
            for (int q = 0; q < 10; ++q) {
                float4 r = row4[q], c = b24[q];
                mx = fmaxf(mx, fmaxf(fmaxf(r.x + c.x, r.y + c.y), fmaxf(r.z + c.z, r.w + c.w)));
            }
            float ssum = 0.0f;
#pragma unroll
            for (int q = 0; q < 10; ++q) {
                float4 r = row4[q], c = b24[q];
                ssum += expf(r.x + c.x - mx) + expf(r.y + c.y - mx)
                      + expf(r.z + c.z - mx) + expf(r.w + c.w - mx);
            }
            float lse = logf(ssum) + mx;
            int n = ii * Mx + jr;
            bool avail = span_avail[n] == 1;
            float4* op = (float4*)(out + ((size_t)n * Bx + b) * OUTx);
#pragma unroll
            for (int q = 0; q < 10; ++q) {
                float4 r = row4[q], c = b24[q], v;
                v.x = avail ? r.x + c.x - lse : -1.0f;
                v.y = avail ? r.y + c.y - lse : -1.0f;
                v.z = avail ? r.z + c.z - lse : -1.0f;
                v.w = avail ? r.w + c.w - lse : -1.0f;
                op[q] = v;
            }
        }
    }
}

// ---- fid head: [cls | masked-mean | fid] @ Wfid + bfid -> log_softmax ----
__global__ void fid_kernel(const float* __restrict__ hidden, const int* __restrict__ pred_spans,
                           const int* __restrict__ token_nums, const float* __restrict__ fid_vecs,
                           const float* __restrict__ Wfid, const float* __restrict__ bfid,
                           float* __restrict__ out) {
    int b = blockIdx.x;
    int t = threadIdx.x;                       // 256 threads
    __shared__ float inp[2 * Hx + FIDx];       // 2236
    __shared__ float red[256];

    int tn = token_nums[b];
    int s = pred_spans[b * 2], e = pred_spans[b * 2 + 1];
    int dis = e - s;
    int lo = tn + 2, hi = tn + 2 + dis;
    if (lo < 0) lo = 0;
    if (hi > Lx - 1) hi = Lx - 1;
    float cnt = (float)(hi - lo + 1);

    for (int h = t; h < Hx; h += 256) {
        inp[h] = hidden[((size_t)b * Lx) * Hx + h];          // cls = hidden[b,0,:]
        float acc = 0.0f;
        for (int l = lo; l <= hi; ++l) acc += hidden[((size_t)(b * Lx + l)) * Hx + h];
        inp[Hx + h] = acc / cnt;
    }
    for (int k = t; k < FIDx; k += 256) inp[2 * Hx + k] = fid_vecs[(size_t)b * FIDx + k];
    __syncthreads();

    float lg[3]; int os[3]; int no = 0;
    for (int o = t; o < FIDx; o += 256) {
        float acc = bfid[o];
        for (int k = 0; k < 2 * Hx + FIDx; ++k) acc += inp[k] * Wfid[(size_t)k * FIDx + o];
        lg[no] = acc; os[no] = o; ++no;
    }
    float mx = -3.4e38f;
    for (int i = 0; i < no; ++i) mx = fmaxf(mx, lg[i]);
    red[t] = mx; __syncthreads();
    for (int st = 128; st > 0; st >>= 1) { if (t < st) red[t] = fmaxf(red[t], red[t + st]); __syncthreads(); }
    mx = red[0]; __syncthreads();
    float ssum = 0.0f;
    for (int i = 0; i < no; ++i) ssum += expf(lg[i] - mx);
    red[t] = ssum; __syncthreads();
    for (int st = 128; st > 0; st >>= 1) { if (t < st) red[t] += red[t + st]; __syncthreads(); }
    float lse = logf(red[0]) + mx;
    float* op = out + (size_t)Mx * Mx * Bx * OUTx + (size_t)b * FIDx;
    for (int i = 0; i < no; ++i) op[os[i]] = lg[i] - lse;
}

extern "C" void kernel_launch(void* const* d_in, const int* in_sizes, int n_in,
                              void* d_out, int out_size, void* d_ws, size_t ws_size,
                              hipStream_t stream) {
    const float* hidden     = (const float*)d_in[0];
    const int*   pred_spans = (const int*)  d_in[1];
    const int*   token_nums = (const int*)  d_in[2];
    const int*   span_avail = (const int*)  d_in[3];
    const float* fid_vecs   = (const float*)d_in[4];
    const float* W1         = (const float*)d_in[5];
    const float* b1         = (const float*)d_in[6];
    const float* W2         = (const float*)d_in[7];
    const float* b2         = (const float*)d_in[8];
    const float* Wfid       = (const float*)d_in[9];
    const float* bfid       = (const float*)d_in[10];
    float* out = (float*)d_out;
    float* ws  = (float*)d_ws;     // needs ~5.9 MB

    prep_kernel<<<(W1F_HALVES + 255) / 256, 256, 0, stream>>>(W1, b1, W2, ws);
    dim3 g1(32, 100);
    aiaj_kernel<<<g1, 32, 0, stream>>>(hidden, ws, ws);
    dim3 g2(16, Mx, Bx);
    span_kernel<<<g2, 32, 0, stream>>>(ws, b2, pred_spans, span_avail, out);
    fid_kernel<<<Bx, 256, 0, stream>>>(hidden, pred_spans, token_nums, fid_vecs, Wfid, bfid, out);
}
